// _Mamba2Block_73495480369507
// MI455X (gfx1250) — matrix operand order repaired and checked by reading, not compiled, not run on hardware
//
#include <hip/hip_runtime.h>
#include <hip/hip_bf16.h>
#include <cstdint>

// ---------------------------------------------------------------------------
// Mamba2-like block for MI455X (gfx1250, wave32, WMMA).
// Pipeline:
//   1) ln_kernel:        LayerNorm(x) -> xn (bf16)
//   2) cvt_bf16_kernel:  in_proj_w, out_proj_w -> bf16
//   3) gemm_bf16_wmma<0>: xz = xn @ in_proj_w^T  (v_wmma_f32_16x16x32_bf16,
//                         async global->LDS staging, prefetch)
//   4) conv_scan_kernel: causal depthwise conv + alpha-scan + silu*gate -> u
//   5) gemm_bf16_wmma<1>: out = u @ out_proj_w^T + residual (fused epilogue)
// ---------------------------------------------------------------------------

#define BB 4
#define SS 4096
#define DD 1024
#define HH 8
#define DI 2048
#define HD 256
#define DCONV 4
#define MM (BB * SS)          // 16384 rows
#define N1 (2 * DI)           // 4096 (in_proj output)
#define LN_EPS 1e-5f

typedef __attribute__((ext_vector_type(8)))  float  v8f;
typedef __attribute__((ext_vector_type(8)))  __bf16 v8bf;
typedef __attribute__((ext_vector_type(16))) __bf16 v16bf;
typedef __attribute__((ext_vector_type(4)))  int    v4i;

__device__ __forceinline__ unsigned short f32_to_bf16(float f) {
    unsigned int u = __float_as_uint(f);
    u = (u + 0x7FFFu + ((u >> 16) & 1u)) >> 16;     // round-to-nearest-even
    return (unsigned short)u;
}
__device__ __forceinline__ float bf16_to_f32(unsigned short h) {
    return __uint_as_float(((unsigned int)h) << 16);
}

// ---------------------------------------------------------------------------
// gfx1250 async global->LDS staging helpers (guarded: falls back cleanly)
// Builtin prototype (from clang diagnostic): (v4i AS1*, v4i AS3*, Ii, Ii)
// ---------------------------------------------------------------------------
#if __has_builtin(__builtin_amdgcn_global_load_async_to_lds_b128)
#define USE_ASYNC_LDS 1
#else
#define USE_ASYNC_LDS 0
#endif

typedef v4i __attribute__((address_space(1))) g_v4i;
typedef v4i __attribute__((address_space(3))) l_v4i;

__device__ __forceinline__ g_v4i* to_global_as(const void* p) {
    // global flat address == AS1 address numerically
    return (g_v4i*)(unsigned long long)(uintptr_t)p;
}
__device__ __forceinline__ l_v4i* to_lds_as(const void* p) {
    // low 32 bits of a flat shared pointer are the LDS byte offset
    return (l_v4i*)(unsigned int)(uintptr_t)p;
}
__device__ __forceinline__ void wait_asynccnt0() {
#if __has_builtin(__builtin_amdgcn_s_wait_asynccnt)
    __builtin_amdgcn_s_wait_asynccnt(0);
#else
    asm volatile("s_wait_asynccnt 0x0" ::: "memory");
#endif
}

// ---------------------------------------------------------------------------
// 1) LayerNorm: one 256-thread block per row (D = 1024, 4 elems/thread)
// ---------------------------------------------------------------------------
__global__ __launch_bounds__(256) void ln_kernel(
    const float* __restrict__ x, const float* __restrict__ nw,
    const float* __restrict__ nb, unsigned short* __restrict__ xn)
{
    const int row = blockIdx.x;
    const float* xr = x + (size_t)row * DD;
    float v[4], s = 0.f, ss = 0.f;
#pragma unroll
    for (int i = 0; i < 4; ++i) {
        float f = xr[threadIdx.x + i * 256];
        v[i] = f; s += f; ss += f * f;
    }
    __shared__ float r0[256], r1[256];
    r0[threadIdx.x] = s; r1[threadIdx.x] = ss;
    __syncthreads();
    for (int str = 128; str > 0; str >>= 1) {
        if ((int)threadIdx.x < str) {
            r0[threadIdx.x] += r0[threadIdx.x + str];
            r1[threadIdx.x] += r1[threadIdx.x + str];
        }
        __syncthreads();
    }
    const float mu  = r0[0] * (1.f / DD);
    const float var = r1[0] * (1.f / DD) - mu * mu;
    const float rs  = rsqrtf(var + LN_EPS);
    unsigned short* o = xn + (size_t)row * DD;
#pragma unroll
    for (int i = 0; i < 4; ++i) {
        int d = threadIdx.x + i * 256;
        o[d] = f32_to_bf16((v[i] - mu) * rs * nw[d] + nb[d]);
    }
}

// ---------------------------------------------------------------------------
// 2) fp32 -> bf16 weight conversion
// ---------------------------------------------------------------------------
__global__ __launch_bounds__(256) void cvt_bf16_kernel(
    const float* __restrict__ in, unsigned short* __restrict__ out, int n)
{
    int i = blockIdx.x * 256 + threadIdx.x;
    if (i < n) out[i] = f32_to_bf16(in[i]);
}

// ---------------------------------------------------------------------------
// 3/5) NT-GEMM  C[M,N] = A[M,K] * W[N,K]^T  via v_wmma_f32_16x16x32_bf16.
//   256 threads = 8 waves (4 x 2). Block tile 128x128, k-step 32.
//   Each wave: 32x64 output = 2x4 WMMA tiles, 8 v8f accumulators.
//   MODE 0: store bf16 to Cbf.  MODE 1: add fp32 residual, store fp32 to Cf.
// ---------------------------------------------------------------------------
#define BM 128
#define BN 128
#define BK 32
#define LDT 40   // LDS row stride in halves (32 + 8 pad; 80 B, 16-B aligned)

template <int MODE>
__global__ __launch_bounds__(256) void gemm_bf16_wmma(
    const unsigned short* __restrict__ A,   // [M, K] bf16
    const unsigned short* __restrict__ W,   // [N, K] bf16
    int Ntot, int Ktot,
    unsigned short* __restrict__ Cbf,
    float* __restrict__ Cf,
    const float* __restrict__ resid)
{
    __shared__ unsigned short As[BM * LDT];
    __shared__ unsigned short Bs[BN * LDT];

    const int tid   = threadIdx.x;
    const int lane  = tid & 31;
    const int wid   = tid >> 5;
    const int wm    = wid & 3;          // 0..3  -> 32-row slab
    const int wn    = wid >> 2;         // 0..1  -> 64-col slab
    const int bm    = blockIdx.y * BM;
    const int bn    = blockIdx.x * BN;
    const int lm    = lane & 15;        // row / col within 16
    const int khalf = lane >> 4;        // K-half select: lane half holds k = 16*khalf .. +15

    v8f acc[2][4] = {};

    for (int k0 = 0; k0 < Ktot; k0 += BK) {
        // --- stage A and W tiles into LDS (16-B chunks, 2 per thread each) ---
#pragma unroll
        for (int i = 0; i < 2; ++i) {
            int linear = tid + i * 256;           // 0..511
            int r = linear >> 2, seg = linear & 3;
            const unsigned short* ag = A + (size_t)(bm + r) * Ktot + k0 + seg * 8;
            const unsigned short* wg = W + (size_t)(bn + r) * Ktot + k0 + seg * 8;
#if USE_ASYNC_LDS
            __builtin_amdgcn_global_load_async_to_lds_b128(
                to_global_as(ag), to_lds_as(&As[r * LDT + seg * 8]), 0, 0);
            __builtin_amdgcn_global_load_async_to_lds_b128(
                to_global_as(wg), to_lds_as(&Bs[r * LDT + seg * 8]), 0, 0);
#else
            *(uint4*)(&As[r * LDT + seg * 8]) = *(const uint4*)ag;
            *(uint4*)(&Bs[r * LDT + seg * 8]) = *(const uint4*)wg;
#endif
            if (k0 + BK < Ktot) {                 // prefetch next k-tile
                __builtin_prefetch(ag + BK, 0, 0);
                __builtin_prefetch(wg + BK, 0, 0);
            }
        }
#if USE_ASYNC_LDS
        wait_asynccnt0();
#endif
        __syncthreads();

        // --- build fragments, same contiguous K order for A and B (k = 16*khalf + s) ---
        v16bf afrag[2], bfrag[4];
#pragma unroll
        for (int i = 0; i < 2; ++i) {
            int m = wm * 32 + i * 16 + lm;
            v8bf lo = *(const v8bf*)(&As[m * LDT + khalf * 16]);      // K 0..7 / 16..23
            v8bf hi = *(const v8bf*)(&As[m * LDT + khalf * 16 + 8]);  // K 8..15 / 24..31
            afrag[i] = __builtin_shufflevector(lo, hi,
                0, 1, 2, 3, 4, 5, 6, 7, 8, 9, 10, 11, 12, 13, 14, 15);
        }
#pragma unroll
        for (int j = 0; j < 4; ++j) {
            int n = wn * 64 + j * 16 + lm;
            bfrag[j] = *(const v16bf*)(&Bs[n * LDT + khalf * 16]);    // 16 contiguous K
        }

#pragma unroll
        for (int i = 0; i < 2; ++i)
#pragma unroll
            for (int j = 0; j < 4; ++j)
                acc[i][j] = __builtin_amdgcn_wmma_f32_16x16x32_bf16(
                    false, afrag[i], false, bfrag[j],
                    (short)0, acc[i][j], false, false);
        __syncthreads();
    }

    // --- epilogue: C/D layout: lane>=16 -> M+8; VGPR r -> row+r; N = lane&15 ---
    const int rsub = khalf * 8;
#pragma unroll
    for (int i = 0; i < 2; ++i) {
#pragma unroll
        for (int j = 0; j < 4; ++j) {
            const int mbase = bm + wm * 32 + i * 16 + rsub;
            const int nidx  = bn + wn * 64 + j * 16 + lm;
#pragma unroll
            for (int r = 0; r < 8; ++r) {
                const size_t idx = (size_t)(mbase + r) * Ntot + nidx;
                const float val = acc[i][j][r];
                if constexpr (MODE == 0) {
                    Cbf[idx] = f32_to_bf16(val);
                } else {
                    Cf[idx] = val + resid[idx];
                }
            }
        }
    }
}

// ---------------------------------------------------------------------------
// 4) depthwise causal conv (DCONV=4) + per-head alpha scan + silu*gate
//    One thread per (batch, channel); coalesced across channels at each s.
// ---------------------------------------------------------------------------
__global__ __launch_bounds__(256) void conv_scan_kernel(
    const unsigned short* __restrict__ xz,   // [B, S, 2*DI] bf16
    const float* __restrict__ cw,            // [DI, 1, DCONV]
    const float* __restrict__ cb,            // [DI]
    const float* __restrict__ log_alpha,     // [H]
    unsigned short* __restrict__ u)          // [B, S, DI] bf16
{
    const int b = blockIdx.x >> 3;                       // DI/256 = 8 blocks/batch
    const int c = (blockIdx.x & 7) * 256 + threadIdx.x;  // channel
    const float w0 = cw[c * DCONV + 0], w1 = cw[c * DCONV + 1];
    const float w2 = cw[c * DCONV + 2], w3 = cw[c * DCONV + 3];
    const float bias  = cb[c];
    const float la    = log_alpha[c >> 8];               // c / HD
    const float alpha = 1.f / (1.f + __expf(-la));

    const unsigned short* xp = xz + (size_t)b * SS * (2 * DI) + c;
    const unsigned short* gp = xp + DI;
    unsigned short* up = u + (size_t)b * SS * DI + c;

    float xm1 = 0.f, xm2 = 0.f, xm3 = 0.f, y = 0.f;
    for (int s = 0; s < SS; ++s) {
        const size_t o = (size_t)s * (2 * DI);
        const float xs = bf16_to_f32(xp[o]);
        const float g  = bf16_to_f32(gp[o]);
        const float cv = w3 * xs + w2 * xm1 + w1 * xm2 + w0 * xm3 + bias;
        y = alpha * y + cv;
        const float sil = y / (1.f + __expf(-y));
        up[(size_t)s * DI] = f32_to_bf16(sil * g);
        xm3 = xm2; xm2 = xm1; xm1 = xs;
    }
}

// ---------------------------------------------------------------------------
// Launch
// ---------------------------------------------------------------------------
extern "C" void kernel_launch(void* const* d_in, const int* in_sizes, int n_in,
                              void* d_out, int out_size, void* d_ws, size_t ws_size,
                              hipStream_t stream)
{
    const float* x          = (const float*)d_in[0];
    const float* norm_w     = (const float*)d_in[1];
    const float* norm_b     = (const float*)d_in[2];
    const float* in_proj_w  = (const float*)d_in[3];   // [2*DI, D]
    const float* conv_w     = (const float*)d_in[4];   // [DI, 1, DCONV]
    const float* conv_b     = (const float*)d_in[5];   // [DI]
    const float* out_proj_w = (const float*)d_in[6];   // [D, DI]
    const float* log_alpha  = (const float*)d_in[7];   // [H]
    float* out = (float*)d_out;

    char* ws = (char*)d_ws;
    size_t off = 0;
    unsigned short* xn  = (unsigned short*)(ws + off); off += (size_t)MM * DD * 2;   // 32 MB
    unsigned short* w1b = (unsigned short*)(ws + off); off += (size_t)N1 * DD * 2;   //  8 MB
    unsigned short* w2b = (unsigned short*)(ws + off); off += (size_t)DD * DI * 2;   //  4 MB
    unsigned short* xzb = (unsigned short*)(ws + off); off += (size_t)MM * N1 * 2;   // 128 MB
    unsigned short* ub  = (unsigned short*)(ws + off); off += (size_t)MM * DI * 2;   // 64 MB

    // weight conversion
    {
        int n1 = N1 * DD;
        cvt_bf16_kernel<<<(n1 + 255) / 256, 256, 0, stream>>>(in_proj_w, w1b, n1);
        int n2 = DD * DI;
        cvt_bf16_kernel<<<(n2 + 255) / 256, 256, 0, stream>>>(out_proj_w, w2b, n2);
    }

    // LayerNorm -> bf16
    ln_kernel<<<MM, 256, 0, stream>>>(x, norm_w, norm_b, xn);

    // GEMM1: xz = xn @ in_proj_w^T   (M=16384, N=4096, K=1024)
    {
        dim3 grid(N1 / BN, MM / BM);   // (32, 128)
        gemm_bf16_wmma<0><<<grid, 256, 0, stream>>>(xn, w1b, N1, DD,
                                                    xzb, nullptr, nullptr);
    }

    // conv + scan + silu*gate -> u (bf16)
    conv_scan_kernel<<<BB * (DI / 256), 256, 0, stream>>>(xzb, conv_w, conv_b,
                                                          log_alpha, ub);

    // GEMM2: out = u @ out_proj_w^T + residual   (M=16384, N=1024, K=2048)
    {
        dim3 grid(DD / BN, MM / BM);   // (8, 128)
        gemm_bf16_wmma<1><<<grid, 256, 0, stream>>>(ub, w2b, DD, DI,
                                                    nullptr, out, x);
    }
}
